// DeepSpeedSelfAttention_77189152243984
// MI455X (gfx1250) — matrix operand order repaired and checked by reading, not compiled, not run on hardware
//
#include <hip/hip_runtime.h>

typedef __bf16 bf16;
typedef __attribute__((ext_vector_type(8)))  __bf16 v8bf;
typedef __attribute__((ext_vector_type(16))) __bf16 v16bf;
typedef __attribute__((ext_vector_type(8)))  float  v8f;
typedef __attribute__((ext_vector_type(4)))  int    v4i;

#define HID   4096
#define SEQ   2048
#define NBAT  2
#define NHEAD 32
#define HDIM  128
#define TOK   (NBAT*SEQ)          // 4096 tokens
#define CH    16777216ull         // 2*2048*4096 elements per output tensor

// ---- feature probes (compile-safe) ----
#if defined(__has_builtin)
#  if __has_builtin(__builtin_amdgcn_perm)
#    define HAVE_PERM 1
#  else
#    define HAVE_PERM 0
#  endif
#  if __has_builtin(__builtin_amdgcn_global_load_async_to_lds_b128) && \
      __has_builtin(__builtin_amdgcn_s_wait_asynccnt)
#    define USE_ASYNC_LDS 1
#  else
#    define USE_ASYNC_LDS 0
#  endif
#else
#  define HAVE_PERM 0
#  define USE_ASYNC_LDS 0
#endif

// ---------------- helpers ----------------
__device__ __forceinline__ unsigned short bfbits(float f) {
  union { float f; unsigned u; } a; a.f = f;
  return (unsigned short)((a.u + 0x8000u) >> 16);   // round-half-up
}

__device__ __forceinline__ bf16 f2bf(float f) {
  union { unsigned short s; bf16 b; } c; c.s = bfbits(f);
  return c.b;
}

// packed f32x2 -> bf16x2: 2 adds + 1 v_perm_b32
__device__ __forceinline__ unsigned pkbf16(float lo, float hi) {
  union { float f; unsigned u; } a, b;
  a.f = lo; b.f = hi;
  const unsigned ul = a.u + 0x8000u, uh = b.u + 0x8000u;
#if HAVE_PERM
  return __builtin_amdgcn_perm(uh, ul, 0x07060302u);  // {uh[31:16], ul[31:16]}
#else
  return (ul >> 16) | (uh & 0xffff0000u);
#endif
}

// pack 16-bit halves selected from two dwords (for bf16-pair transpose)
__device__ __forceinline__ unsigned pick16(unsigned lo_src, unsigned hi_src, int odd) {
#if HAVE_PERM
  return __builtin_amdgcn_perm(hi_src, lo_src, odd ? 0x07060302u : 0x05040100u);
#else
  return odd ? ((lo_src >> 16) | (hi_src & 0xffff0000u))
             : ((lo_src & 0xffffu) | (hi_src << 16));
#endif
}

__device__ __forceinline__ v16bf catbf(v8bf lo, v8bf hi) {
  v16bf r;
#pragma unroll
  for (int i = 0; i < 8; ++i) { r[i] = lo[i]; r[i + 8] = hi[i]; }
  return r;
}

__device__ __forceinline__ v8f zero8() {
  v8f z = {0.f,0.f,0.f,0.f,0.f,0.f,0.f,0.f};
  return z;
}

#if USE_ASYNC_LDS
__device__ __forceinline__ void async_copy16(const bf16* gsrc, bf16* lds_dst) {
  __builtin_amdgcn_global_load_async_to_lds_b128(
      (__attribute__((address_space(1))) v4i*)(void*)gsrc,
      (__attribute__((address_space(3))) v4i*)(void*)lds_dst, 0, 0);
}
#endif

// ---------------- kernel 1: layernorm (+bf16 copy) ----------------
__global__ __launch_bounds__(256)
void ln_kernel(const float* __restrict__ x, const float* __restrict__ w,
               const float* __restrict__ b, float* __restrict__ out_norm,
               bf16* __restrict__ xn_bf) {
  __shared__ float red1[8], red2[8];
  const int t   = blockIdx.x;                 // token
  const int tid = threadIdx.x;
  const int lane = tid & 31, wv = tid >> 5;
  const float* xr = x + (size_t)t * HID;
  float lsum = 0.f, lsq = 0.f;
  for (int i = tid; i < HID; i += 256) { float v = xr[i]; lsum += v; lsq += v * v; }
#pragma unroll
  for (int off = 16; off >= 1; off >>= 1) {
    lsum += __shfl_xor(lsum, off, 32);
    lsq  += __shfl_xor(lsq,  off, 32);
  }
  if (lane == 0) { red1[wv] = lsum; red2[wv] = lsq; }
  __syncthreads();
  float ts = 0.f, tq = 0.f;
#pragma unroll
  for (int i = 0; i < 8; ++i) { ts += red1[i]; tq += red2[i]; }
  const float mu   = ts * (1.f / (float)HID);
  const float var  = tq * (1.f / (float)HID) - mu * mu;
  const float rstd = rsqrtf(var + 1e-5f);
  for (int i = tid; i < HID; i += 256) {
    float nv = (xr[i] - mu) * rstd * w[i] + b[i];
    out_norm[(size_t)t * HID + i] = nv;
    xn_bf[(size_t)t * HID + i]    = f2bf(nv);
  }
}

// ---------------- kernel 2/4: double-buffered bf16 WMMA GEMM ----------------
// C[M=4096, N] = A_bf[4096,4096] * B_f32[4096,N]  (+bias, mode 0 = QKV scatter)
// Grid: x = M tile (fastest -> B panel stays L2-resident), y = N tile.
__global__ __launch_bounds__(256)
void gemm_bf16_kernel(const bf16* __restrict__ A, const float* __restrict__ Bm,
                      const float* __restrict__ bias, int N, int mode,
                      float* __restrict__ out_k, float* __restrict__ out_v,
                      bf16* __restrict__ qbf, bf16* __restrict__ kbf,
                      bf16* __restrict__ vbf, float* __restrict__ out_plain) {
  __shared__ __align__(16) bf16 As[2][128][40];   // [buf][row][k]
  __shared__ __align__(16) bf16 Bs[2][128][40];   // [buf][n][k]  (B transposed)
  const int tid  = threadIdx.x;
  const int lane = tid & 31;
  const int wv   = tid >> 5;
  const int half = lane >> 4;
  const int l15  = lane & 15;
  const int rowBase = blockIdx.x * 128;        // M varies fastest across blocks
  const int colBase = blockIdx.y * 128;
  const int waveRow = (wv >> 1) * 32;          // 4x2 wave grid, 32x64 wave tile
  const int waveCol = (wv & 1) * 64;

  // --- per-thread staging coordinates (fixed for whole kernel) ---
  const int c0 = tid * 2, c1 = c0 + 1;
  const int rA0 = c0 >> 2, cA0 = (c0 & 3) * 8;    // A: 128 rows x 32 k, 16B chunks
  const int rA1 = c1 >> 2, cA1 = (c1 & 3) * 8;
  const int kB0 = (c0 >> 5) * 2, nB0 = (c0 & 31) * 4;  // B: 2k x 4n micro-tiles
  const int kB1 = (c1 >> 5) * 2, nB1 = (c1 & 31) * 4;

  const bf16*  pa0 = A  + (size_t)(rowBase + rA0) * HID + cA0;
  const bf16*  pa1 = A  + (size_t)(rowBase + rA1) * HID + cA1;
  const float* pb0 = Bm + (size_t)kB0 * N + colBase + nB0;
  const float* pb1 = Bm + (size_t)kB1 * N + colBase + nB1;

  auto store_stage_b = [&](int p, float4 b00, float4 b01, float4 b10, float4 b11) {
    const float x0[4] = {b00.x, b00.y, b00.z, b00.w};
    const float x1[4] = {b01.x, b01.y, b01.z, b01.w};
    const float y0[4] = {b10.x, b10.y, b10.z, b10.w};
    const float y1[4] = {b11.x, b11.y, b11.z, b11.w};
#pragma unroll
    for (int i = 0; i < 4; ++i)
      *(unsigned*)&Bs[p][nB0 + i][kB0] = pkbf16(x0[i], x1[i]);
#pragma unroll
    for (int i = 0; i < 4; ++i)
      *(unsigned*)&Bs[p][nB1 + i][kB1] = pkbf16(y0[i], y1[i]);
  };

  v8f acc[2][4];
#pragma unroll
  for (int mt = 0; mt < 2; ++mt)
#pragma unroll
    for (int nt = 0; nt < 4; ++nt) acc[mt][nt] = zero8();

  // --- prologue: stage k0 = 0 into buffer 0 ---
  {
#if USE_ASYNC_LDS
    async_copy16(pa0, &As[0][rA0][cA0]);
    async_copy16(pa1, &As[0][rA1][cA1]);
#else
    v8bf ra0 = *(const v8bf*)pa0, ra1 = *(const v8bf*)pa1;
    *(v8bf*)&As[0][rA0][cA0] = ra0;
    *(v8bf*)&As[0][rA1][cA1] = ra1;
#endif
    pa0 += 32; pa1 += 32;
    float4 b00 = *(const float4*)pb0, b01 = *(const float4*)(pb0 + N);
    float4 b10 = *(const float4*)pb1, b11 = *(const float4*)(pb1 + N);
    pb0 += (size_t)32 * N; pb1 += (size_t)32 * N;
    store_stage_b(0, b00, b01, b10, b11);
#if USE_ASYNC_LDS
    __builtin_amdgcn_s_wait_asynccnt(0);
#endif
  }
  __syncthreads();

#pragma unroll 1
  for (int k0 = 0; k0 < HID; k0 += 32) {
    const int p = (k0 >> 5) & 1;
    const bool more = (k0 + 32) < HID;

    // issue next tile's transfers early (hide latency under WMMA)
#if !USE_ASYNC_LDS
    v8bf  ra0{}, ra1{};
#endif
    float4 b00{}, b01{}, b10{}, b11{};
    if (more) {
#if USE_ASYNC_LDS
      async_copy16(pa0, &As[p ^ 1][rA0][cA0]);     // DMA straight into LDS
      async_copy16(pa1, &As[p ^ 1][rA1][cA1]);
#else
      __builtin_prefetch((const void*)(pa0 + 32), 0, 1);
      ra0 = *(const v8bf*)pa0; ra1 = *(const v8bf*)pa1;
#endif
      pa0 += 32; pa1 += 32;
      b00 = *(const float4*)pb0; b01 = *(const float4*)(pb0 + N);
      b10 = *(const float4*)pb1; b11 = *(const float4*)(pb1 + N);
      pb0 += (size_t)32 * N; pb1 += (size_t)32 * N;
    }

    // compute from buffer p (CDNA5 wave32 fragment layouts)
    v16bf afrag[2], bfrag[4];
#pragma unroll
    for (int mt = 0; mt < 2; ++mt) {
      const int r = waveRow + mt * 16 + l15;
      v8bf lo = *(const v8bf*)&As[p][r][8 * half];        // K = 8h..8h+7
      v8bf hi = *(const v8bf*)&As[p][r][16 + 8 * half];   // K = 16+8h..
      afrag[mt] = catbf(lo, hi);
    }
#pragma unroll
    for (int nt = 0; nt < 4; ++nt) {
      const int n = waveCol + nt * 16 + l15;
      v8bf lo = *(const v8bf*)&Bs[p][n][8 * half];        // K = 8h..8h+7
      v8bf hi = *(const v8bf*)&Bs[p][n][16 + 8 * half];   // K = 16+8h..
      bfrag[nt] = catbf(lo, hi);
    }
#pragma unroll
    for (int mt = 0; mt < 2; ++mt)
#pragma unroll
      for (int nt = 0; nt < 4; ++nt)
        acc[mt][nt] = __builtin_amdgcn_wmma_f32_16x16x32_bf16(
            false, afrag[mt], false, bfrag[nt], (short)0, acc[mt][nt], false, false);

    if (more) {
      store_stage_b(p ^ 1, b00, b01, b10, b11);
#if !USE_ASYNC_LDS
      *(v8bf*)&As[p ^ 1][rA0][cA0] = ra0;
      *(v8bf*)&As[p ^ 1][rA1][cA1] = ra1;
#endif
    }
#if USE_ASYNC_LDS
    if (more) __builtin_amdgcn_s_wait_asynccnt(0);
#endif
    __syncthreads();
  }

  // epilogue
#pragma unroll
  for (int mt = 0; mt < 2; ++mt) {
#pragma unroll
    for (int nt = 0; nt < 4; ++nt) {
#pragma unroll
      for (int v = 0; v < 8; ++v) {
        const int row = rowBase + waveRow + mt * 16 + half * 8 + v;
        const int col = colBase + waveCol + nt * 16 + l15;
        float val = acc[mt][nt][v];
        if (mode == 0) {
          val += bias[col];
          const int part = col >> 12;           // 0=q 1=k 2=v
          const int cc   = col & 4095;
          const int head = cc >> 7, d = cc & 127;
          const int bb   = row >> 11, s = row & 2047;
          const size_t idx = ((size_t)((bb * NHEAD + head) * SEQ + s)) * HDIM + d;
          if (part == 0)      { qbf[idx] = f2bf(val); }
          else if (part == 1) { out_k[idx] = val; kbf[idx] = f2bf(val); }
          else                { out_v[idx] = val; vbf[idx] = f2bf(val); }
        } else {
          out_plain[(size_t)row * HID + col] = val;
        }
      }
    }
  }
}

// ---------------- kernel 3: flash-style attention ----------------
__global__ __launch_bounds__(256)
void attn_kernel(const bf16* __restrict__ qbf, const bf16* __restrict__ kbf,
                 const bf16* __restrict__ vbf, const float* __restrict__ mask,
                 float* __restrict__ ctx_out, bf16* __restrict__ ctx_bf) {
  __shared__ __align__(16) bf16 Vt[128][40];       // V chunk transposed [d][tok]
  __shared__ __align__(16) bf16 Pw[8][16][40];     // per-wave probs 16x32
  const int tid  = threadIdx.x;
  const int lane = tid & 31;
  const int wv   = tid >> 5;
  const int half = lane >> 4;
  const int l15  = lane & 15;
  const int bx = blockIdx.x;                       // q tile (0..15)
  const int by = blockIdx.y;                       // b*32+h (0..63)
  const int bb = by >> 5;
  const int hh = by & 31;
  const size_t headOff = (size_t)by * SEQ * HDIM;
  const float scale = 0.08838834764831845f;        // 1/sqrt(128)

  // V staging coords: one 2tok x 8d unit per thread
  const int tokp = tid >> 4;          // 0..15 (token pair)
  const int vdc  = (tid & 15) * 8;    // d chunk

  // Q fragments: global layout already matches A-fragment layout
  const int qRow = bx * 128 + wv * 16 + l15;
  v16bf aq[4];
#pragma unroll
  for (int kc = 0; kc < 4; ++kc) {
    const bf16* qr = qbf + headOff + (size_t)qRow * HDIM + kc * 32;
    v8bf lo = *(const v8bf*)(qr + 8 * half);
    v8bf hi = *(const v8bf*)(qr + 16 + 8 * half);
    aq[kc] = catbf(lo, hi);
  }

  v8f acc[8];
#pragma unroll
  for (int nt = 0; nt < 8; ++nt) acc[nt] = zero8();
  float m[8], ssum[8];
#pragma unroll
  for (int v = 0; v < 8; ++v) { m[v] = -1e30f; ssum[v] = 0.f; }

#pragma unroll 1
  for (int tc = 0; tc < SEQ / 32; ++tc) {
    const int t0 = tc * 32;
    // stage V chunk transposed (token-pair packs via v_perm_b32)
    {
      const bf16* vsrc = vbf + headOff + (size_t)(t0 + 2 * tokp) * HDIM + vdc;
      __builtin_prefetch((const void*)(vsrc + 32 * HDIM), 0, 1);
      union { v8bf v; unsigned u[4]; } U0, U1;
      U0.v = *(const v8bf*)vsrc;
      U1.v = *(const v8bf*)(vsrc + HDIM);
#pragma unroll
      for (int e = 0; e < 8; ++e)
        *(unsigned*)&Vt[vdc + e][2 * tokp] =
            pick16(U0.u[e >> 1], U1.u[e >> 1], e & 1);
    }
    __syncthreads();

    // scores = Q * K^T for 2 x (16x16) tiles
    v8f sc[2];
#pragma unroll
    for (int st = 0; st < 2; ++st) {
      sc[st] = zero8();
      const int n = t0 + st * 16 + l15;
      const bf16* krow = kbf + headOff + (size_t)n * HDIM;
      __builtin_prefetch((const void*)(krow + 32 * HDIM), 0, 1);
#pragma unroll
      for (int kc = 0; kc < 4; ++kc) {
        v8bf lo = *(const v8bf*)(krow + kc * 32 + 8 * half);
        v8bf hi = *(const v8bf*)(krow + kc * 32 + 16 + 8 * half);
        sc[st] = __builtin_amdgcn_wmma_f32_16x16x32_bf16(
            false, aq[kc], false, catbf(lo, hi), (short)0, sc[st], false, false);
      }
      const float mval = mask[bb * SEQ + t0 + st * 16 + l15];
#pragma unroll
      for (int v = 0; v < 8; ++v) sc[st][v] = sc[st][v] * scale + mval;
    }

    // online softmax (row reductions across the 16 lanes of each half)
    float nm[8], fct[8];
#pragma unroll
    for (int v = 0; v < 8; ++v) {
      float t = fmaxf(sc[0][v], sc[1][v]);
#pragma unroll
      for (int off = 1; off < 16; off <<= 1) t = fmaxf(t, __shfl_xor(t, off, 32));
      nm[v] = fmaxf(m[v], t);
      sc[0][v] = __expf(sc[0][v] - nm[v]);
      sc[1][v] = __expf(sc[1][v] - nm[v]);
      float r = sc[0][v] + sc[1][v];
#pragma unroll
      for (int off = 1; off < 16; off <<= 1) r += __shfl_xor(r, off, 32);
      fct[v] = __expf(m[v] - nm[v]);
      ssum[v] = ssum[v] * fct[v] + r;
      m[v] = nm[v];
    }
#pragma unroll
    for (int nt = 0; nt < 8; ++nt)
#pragma unroll
      for (int v = 0; v < 8; ++v) acc[nt][v] *= fct[v];

    // C-layout -> A-layout for probs via wave-private LDS
#pragma unroll
    for (int st = 0; st < 2; ++st)
#pragma unroll
      for (int v = 0; v < 8; ++v)
        Pw[wv][half * 8 + v][st * 16 + l15] = f2bf(sc[st][v]);
    v8bf plo = *(const v8bf*)&Pw[wv][l15][8 * half];
    v8bf phi = *(const v8bf*)&Pw[wv][l15][16 + 8 * half];
    v16bf ap = catbf(plo, phi);

    // ctx += P * V
#pragma unroll
    for (int nt = 0; nt < 8; ++nt) {
      const int d = nt * 16 + l15;
      v8bf blo = *(const v8bf*)&Vt[d][8 * half];
      v8bf bhi = *(const v8bf*)&Vt[d][16 + 8 * half];
      acc[nt] = __builtin_amdgcn_wmma_f32_16x16x32_bf16(
          false, ap, false, catbf(blo, bhi), (short)0, acc[nt], false, false);
    }
    __syncthreads();
  }

  // epilogue: normalize, write ctx (fp32 out + bf16 ws) in [B,S,H] layout
  float inv[8];
#pragma unroll
  for (int v = 0; v < 8; ++v) inv[v] = 1.f / ssum[v];
  const int sBase = bx * 128 + wv * 16;
#pragma unroll
  for (int nt = 0; nt < 8; ++nt) {
#pragma unroll
    for (int v = 0; v < 8; ++v) {
      const int s = sBase + half * 8 + v;
      const int d = nt * 16 + l15;
      const float val = acc[nt][v] * inv[v];
      const size_t o = ((size_t)(bb * SEQ + s)) * HID + hh * HDIM + d;
      ctx_out[o] = val;
      ctx_bf[o]  = f2bf(val);
    }
  }
}

// ---------------- launch ----------------
extern "C" void kernel_launch(void* const* d_in, const int* in_sizes, int n_in,
                              void* d_out, int out_size, void* d_ws, size_t ws_size,
                              hipStream_t stream) {
  (void)in_sizes; (void)n_in; (void)out_size; (void)ws_size;
  const float* x     = (const float*)d_in[0];
  const float* mask  = (const float*)d_in[1];
  const float* nw    = (const float*)d_in[2];
  const float* nb    = (const float*)d_in[3];
  const float* qkvw  = (const float*)d_in[4];
  const float* qkvb  = (const float*)d_in[5];
  const float* ow    = (const float*)d_in[6];

  float* out      = (float*)d_out;
  float* out_o    = out;            // [B,S,H] attention output
  float* out_k    = out + CH;       // [B,heads,S,hd]
  float* out_v    = out + 2 * CH;   // [B,heads,S,hd]
  float* out_ctx  = out + 3 * CH;   // [B,S,H]
  float* out_norm = out + 4 * CH;   // [B,S,H]

  bf16* wsb = (bf16*)d_ws;
  bf16* xn  = wsb;                  // layernorm output bf16
  bf16* qb  = wsb + CH;             // q bf16 [B,h,S,hd]
  bf16* kb  = wsb + 2 * CH;
  bf16* vb  = wsb + 3 * CH;
  bf16* cb  = wsb + 4 * CH;         // ctx bf16 [B,S,H]

  ln_kernel<<<TOK, 256, 0, stream>>>(x, nw, nb, out_norm, xn);
  // grid.x = M tiles (fastest) so the 2MB B panel is reused out of L2
  gemm_bf16_kernel<<<dim3(32, 96), 256, 0, stream>>>(
      xn, qkvw, qkvb, 3 * HID, 0, out_k, out_v, qb, kb, vb, nullptr);
  attn_kernel<<<dim3(SEQ / 128, NBAT * NHEAD), 256, 0, stream>>>(
      qb, kb, vb, mask, out_ctx, cb);
  gemm_bf16_kernel<<<dim3(32, 32), 256, 0, stream>>>(
      cb, ow, nullptr, HID, 1, nullptr, nullptr, nullptr, nullptr, nullptr, out_o);
}
